// WindowAttentionUp_25159918420293
// MI455X (gfx1250) — matrix operand order repaired and checked by reading, not compiled, not run on hardware
//
#include <hip/hip_runtime.h>

// ---------------------------------------------------------------------------
// SwinV2 window attention for MI455X (gfx1250), bf16 WMMA path.
// ---------------------------------------------------------------------------

typedef __attribute__((ext_vector_type(16))) __bf16 v16bf;
typedef __attribute__((ext_vector_type(8)))  float  v8f;

#define DIM   256
#define NTOK  64
#define HEADS 8
#define HD    32
#define NWIN  64
#define BWIN  2048

static __device__ __forceinline__ v8f wmma_bf16(v16bf a, v16bf b, v8f c) {
  return __builtin_amdgcn_wmma_f32_16x16x32_bf16(false, a, false, b,
                                                 (short)0, c, false, false);
}

// Branch-free round-to-nearest-even fp32 -> bf16 (data is NaN-free here).
static __device__ __forceinline__ unsigned bf16_bits(float f) {
  unsigned u = __float_as_uint(f);
  return (u + 0x7FFFu + ((u >> 16) & 1u)) >> 16;
}
static __device__ __forceinline__ unsigned pack_bf16(float lo, float hi) {
  return bf16_bits(lo) | (bf16_bits(hi) << 16);
}
static __device__ __forceinline__ __bf16 to_bf16(float f) {
  unsigned short s = (unsigned short)bf16_bits(f);
  return __builtin_bit_cast(__bf16, s);
}

struct U4x2 { uint4 a, b; };

// A fragment (16x32 row-major, stride rs elems; rows must be 16B-aligned).
// ISA: lane<16 row=lane K={0..7,16..23}; lane>=16 K={8..15,24..31}.
static __device__ __forceinline__ v16bf load_a_frag(const __bf16* base, int rs, int lane) {
  const int r  = lane & 15;
  const int kb = (lane & 16) ? 8 : 0;
  const __bf16* p = base + r * rs + kb;
  U4x2 t;
  t.a = *(const uint4*)(p);        // K = kb .. kb+7
  t.b = *(const uint4*)(p + 16);   // K = kb+16 .. kb+23
  return __builtin_bit_cast(v16bf, t);
}

// B fragment when memory holds B^T row-major (N rows x K cols, stride rs).
// Matches A order: lanes 0-15 hold K={0..7,16..23} of col n=lane; lanes 16-31 K={8..15,24..31}.
static __device__ __forceinline__ v16bf load_bt_frag(const __bf16* base, int rs, int lane) {
  const int n  = lane & 15;
  const int kb = (lane & 16) ? 8 : 0;
  const __bf16* p = base + n * rs + kb;
  U4x2 t;
  t.a = *(const uint4*)(p);
  t.b = *(const uint4*)(p + 16);
  return __builtin_bit_cast(v16bf, t);
}

// ---------------------------------------------------------------------------
// Kernel 1a: CPB bias MLP (2->256->8) + 1/max(tau,0.01) -> interleaved float2
// ---------------------------------------------------------------------------
__global__ __launch_bounds__(256) void prep_kernel(
    const float* __restrict__ w1, const float* __restrict__ b1,
    const float* __restrict__ w2, const float* __restrict__ b2,
    const float* __restrict__ tau, const float* __restrict__ lri,
    float2* __restrict__ combo)
{
  const int p = blockIdx.x * 256 + threadIdx.x;  // 0..4095
  const float l0 = lri[p * 2 + 0];
  const float l1 = lri[p * 2 + 1];
  float acc[HEADS] = {};
  for (int j = 0; j < 256; ++j) {
    float hv = fmaf(l0, w1[j], fmaf(l1, w1[256 + j], b1[j]));
    hv = fmaxf(hv, 0.0f);
#pragma unroll
    for (int h = 0; h < HEADS; ++h) acc[h] = fmaf(hv, w2[j * HEADS + h], acc[h]);
  }
#pragma unroll
  for (int h = 0; h < HEADS; ++h) {
    float2 c;
    c.x = __builtin_amdgcn_rcpf(fmaxf(tau[h * 4096 + p], 0.01f));
    c.y = acc[h] + b2[h];
    combo[h * 4096 + p] = c;
  }
}

// ---------------------------------------------------------------------------
// Kernel 1b: transpose weights fp32 [K=256][N] -> bf16 [N][K=256] (one-time).
// ---------------------------------------------------------------------------
__global__ __launch_bounds__(256) void wtrans_kernel(
    const float* __restrict__ W, __bf16* __restrict__ Wt, int Ncols)
{
  const int idx = blockIdx.x * 256 + threadIdx.x;  // over Ncols*256
  const int n = idx >> 8, k = idx & 255;
  Wt[idx] = to_bf16(W[(size_t)k * Ncols + n]);
}

// ---------------------------------------------------------------------------
// Kernel 2/3/5: tiled GEMM  C[M x Ncols] = A[M x 256] @ W + bias
// B fragments come straight from the pre-transposed bf16 weights (L2-hot).
// MODE 0: Q bf16 [b][h][n][d]; MODE 1: K [b][h][n][d] + V [b][h][d][n];
// MODE 2: fp32 row-major (final output).
// ---------------------------------------------------------------------------
template<int MODE, bool ABF16>
__global__ __launch_bounds__(256) void gemm_bf16_kernel(
    const float* __restrict__ Af, const __bf16* __restrict__ Ab,
    const __bf16* __restrict__ Wt,  // [Ncols][256] bf16
    const float* __restrict__ bias,
    __bf16* __restrict__ out0,   // MODE0: q, MODE1: k
    __bf16* __restrict__ out1,   // MODE1: v
    float*  __restrict__ fout)   // MODE2
{
  __shared__ __bf16 As[128][40];   // 128x32 A tile (80B rows, 16B aligned)
  const int tid  = threadIdx.x;
  const int n0   = blockIdx.x * 64;
  const int m0   = blockIdx.y * 128;
  const int wave = tid >> 5, lane = tid & 31;
  const int wm   = (wave & 3) * 32;   // 4 waves tile M
  const int wn   = (wave >> 2) * 32;  // 2 waves tile N
  const int arow = tid >> 1, aseg = (tid & 1) * 16;

  const __bf16* wt0 = Wt + (size_t)(n0 + wn) * DIM;
  const __bf16* wt1 = Wt + (size_t)(n0 + wn + 16) * DIM;

  v8f acc[2][2] = {};

  for (int kc = 0; kc < DIM; kc += 32) {
    if (ABF16) {
      const uint4* ap4 = (const uint4*)(Ab + (size_t)(m0 + arow) * DIM + kc + aseg);
      *(uint4*)&As[arow][aseg]     = ap4[0];
      *(uint4*)&As[arow][aseg + 8] = ap4[1];
    } else {
      const float4* ap4 = (const float4*)(Af + (size_t)(m0 + arow) * DIM + kc + aseg);
      float4 f0 = ap4[0], f1 = ap4[1], f2 = ap4[2], f3 = ap4[3];
      *(uint4*)&As[arow][aseg] =
          make_uint4(pack_bf16(f0.x, f0.y), pack_bf16(f0.z, f0.w),
                     pack_bf16(f1.x, f1.y), pack_bf16(f1.z, f1.w));
      *(uint4*)&As[arow][aseg + 8] =
          make_uint4(pack_bf16(f2.x, f2.y), pack_bf16(f2.z, f2.w),
                     pack_bf16(f3.x, f3.y), pack_bf16(f3.z, f3.w));
    }
    __syncthreads();

    v16bf a0 = load_a_frag(&As[wm][0],      40, lane);
    v16bf a1 = load_a_frag(&As[wm + 16][0], 40, lane);
    v16bf b0 = load_bt_frag(wt0 + kc, DIM, lane);
    v16bf b1 = load_bt_frag(wt1 + kc, DIM, lane);
    acc[0][0] = wmma_bf16(a0, b0, acc[0][0]);
    acc[0][1] = wmma_bf16(a0, b1, acc[0][1]);
    acc[1][0] = wmma_bf16(a1, b0, acc[1][0]);
    acc[1][1] = wmma_bf16(a1, b1, acc[1][1]);
    __syncthreads();
  }

  const int coloff = lane & 15;
  const int rowoff = (lane & 16) ? 8 : 0;
#pragma unroll
  for (int ti = 0; ti < 2; ++ti) {
#pragma unroll
    for (int tj = 0; tj < 2; ++tj) {
      const int Ng = n0 + wn + tj * 16 + coloff;   // constant per tile
      const float bv = bias[Ng];
      const int Mb = m0 + wm + ti * 16 + rowoff;   // multiple of 8
      if (MODE == 2) {
        float* fb = fout + (size_t)Mb * DIM + Ng;
#pragma unroll
        for (int r = 0; r < 8; ++r) fb[r * DIM] = acc[ti][tj][r] + bv;
      } else {
        const int b = Mb >> 6, nw = Mb & 63;       // constant per tile (8-aligned)
        if (MODE == 0 || Ng < DIM) {
          const int h = Ng >> 5, d = Ng & 31;
          __bf16* qp = out0 + (((size_t)b * HEADS + h) * NTOK + nw) * HD + d;
#pragma unroll
          for (int r = 0; r < 8; ++r) qp[r * HD] = to_bf16(acc[ti][tj][r] + bv);
        } else {
          const int c2 = Ng - DIM, h = c2 >> 5, d = c2 & 31;
          __bf16* vp = out1 + (((size_t)b * HEADS + h) * HD + d) * NTOK + nw;
#pragma unroll
          for (int r = 0; r < 8; r += 2)           // consecutive tokens: pack b32
            *(unsigned*)(vp + r) =
                pack_bf16(acc[ti][tj][r] + bv, acc[ti][tj][r + 1] + bv);
        }
      }
    }
  }
}

// ---------------------------------------------------------------------------
// Kernel 4: fused per-window cosine attention.
// grid = 2 blocks per window (4 heads each), 128 threads (1 head per wave).
// ---------------------------------------------------------------------------
__global__ __launch_bounds__(128) void attn_kernel(
    const __bf16* __restrict__ qb, const __bf16* __restrict__ kb,
    const __bf16* __restrict__ vb, const float* __restrict__ mask,
    const float2* __restrict__ combo, __bf16* __restrict__ aout)
{
  __shared__ float  mask_s[NTOK * NTOK];   // 16 KB, filled via async-to-LDS
  __shared__ float  rqn_s[4][NTOK];        // inverse q norms
  __shared__ float  rkn_s[4][NTOK];        // inverse k norms
  __shared__ __bf16 P_s[4][NTOK][72];      // probs; 144B rows, 16B aligned

  const int b    = blockIdx.x >> 1;
  const int wv   = threadIdx.x >> 5;
  const int h    = ((blockIdx.x & 1) * 4) + wv;
  const int lane = threadIdx.x & 31;

  // Stage the window mask with CDNA5 async global->LDS copies (ASYNCcnt).
  {
    const float* mp = mask + (size_t)(b & (NWIN - 1)) * NTOK * NTOK;
#pragma unroll
    for (int i = 0; i < 8; ++i) {
      const int idx = i * 128 + threadIdx.x;          // 16-byte chunk index
      unsigned lds = (unsigned)(unsigned long long)(&mask_s[idx * 4]);
      unsigned long long ga = (unsigned long long)(mp + idx * 4);
      asm volatile("global_load_async_to_lds_b128 %0, %1, off"
                   :: "v"(lds), "v"(ga) : "memory");
    }
  }

  const __bf16* qh = qb + ((size_t)b * HEADS + h) * NTOK * HD;  // [n][d]
  const __bf16* kh = kb + ((size_t)b * HEADS + h) * NTOK * HD;  // [n][d]
  const __bf16* vh = vb + ((size_t)b * HEADS + h) * HD * NTOK;  // [d][n]

  v16bf qa[4], ka[4];
#pragma unroll
  for (int mi = 0; mi < 4; ++mi) qa[mi] = load_a_frag(qh + mi * 16 * HD, HD, lane);
#pragma unroll
  for (int ni = 0; ni < 4; ++ni) ka[ni] = load_bt_frag(kh + ni * 16 * HD, HD, lane);

  // Inverse norms via hardware v_rsq_f32. Lanes L and L+16 hold the same
  // reduced value and write the same address: unconditional store (no exec
  // mask juggling; benign same-value write).
#pragma unroll
  for (int mi = 0; mi < 4; ++mi) {
    float s = 0.0f;
#pragma unroll
    for (int e = 0; e < 16; ++e) { float f = (float)qa[mi][e]; s += f * f; }
    s += __shfl_xor(s, 16, 32);
    rqn_s[wv][mi * 16 + (lane & 15)] = __builtin_amdgcn_rsqf(s);
  }
#pragma unroll
  for (int ni = 0; ni < 4; ++ni) {
    float s = 0.0f;
#pragma unroll
    for (int e = 0; e < 16; ++e) { float f = (float)ka[ni][e]; s += f * f; }
    s += __shfl_xor(s, 16, 32);
    rkn_s[wv][ni * 16 + (lane & 15)] = __builtin_amdgcn_rsqf(s);
  }
  asm volatile("s_wait_asynccnt 0" ::: "memory");
  __syncthreads();

  // S = q . k^T (64x64 per head, 16 WMMAs)
  v8f S[4][4] = {};
#pragma unroll
  for (int mi = 0; mi < 4; ++mi)
#pragma unroll
    for (int ni = 0; ni < 4; ++ni)
      S[mi][ni] = wmma_bf16(qa[mi], ka[ni], S[mi][ni]);

  const int coloff = lane & 15;
  const int rowoff = (lane & 16) ? 8 : 0;
  // One per-lane base; remaining offsets are compile-time constants.
  const float2* cb  = combo + (size_t)h * 4096 + rowoff * 64 + coloff;
  const float*  mb  = &mask_s[rowoff * 64 + coloff];
  const float*  qnb = &rqn_s[wv][rowoff];
  const float*  knb = &rkn_s[wv][coloff];
#pragma unroll
  for (int mi = 0; mi < 4; ++mi)
#pragma unroll
    for (int r = 0; r < 8; ++r) {
      const float rq = qnb[mi * 16 + r];
#pragma unroll
      for (int ni = 0; ni < 4; ++ni) {
        const int idx = (mi * 16 + r) * 64 + ni * 16;
        const float2 c = cb[idx];
        // num / max(qn*kn, 1e-6)  ==  num * min(rq*rk, 1e6)
        const float invdn = fminf(rq * knb[ni * 16], 1e6f);
        S[mi][ni][r] = fmaf(S[mi][ni][r] * invdn, c.x, c.y + mb[idx]);
      }
    }

  // Row softmax: row lives in one 16-lane half-group x 4 ni registers.
#pragma unroll
  for (int mi = 0; mi < 4; ++mi)
#pragma unroll
    for (int r = 0; r < 8; ++r) {
      float mx = -3.4e38f;
#pragma unroll
      for (int ni = 0; ni < 4; ++ni) mx = fmaxf(mx, S[mi][ni][r]);
#pragma unroll
      for (int off = 8; off >= 1; off >>= 1) mx = fmaxf(mx, __shfl_xor(mx, off, 32));
      float sum = 0.0f;
#pragma unroll
      for (int ni = 0; ni < 4; ++ni) {
        float e = __expf(S[mi][ni][r] - mx);
        S[mi][ni][r] = e; sum += e;
      }
#pragma unroll
      for (int off = 8; off >= 1; off >>= 1) sum += __shfl_xor(sum, off, 32);
      const float inv = __builtin_amdgcn_rcpf(sum);   // v_rcp_f32
#pragma unroll
      for (int ni = 0; ni < 4; ++ni) S[mi][ni][r] *= inv;
    }

  // Stage probabilities for the P@V A-fragments.
  {
    __bf16* pb = &P_s[wv][rowoff][coloff];
#pragma unroll
    for (int mi = 0; mi < 4; ++mi)
#pragma unroll
      for (int ni = 0; ni < 4; ++ni)
#pragma unroll
        for (int r = 0; r < 8; ++r)
          pb[(mi * 16 + r) * 72 + ni * 16] = to_bf16(S[mi][ni][r]);
  }
  __syncthreads();

  // O = P @ V (64x32 per head, 16 WMMAs)
  v8f O[4][2] = {};
#pragma unroll
  for (int kc = 0; kc < 2; ++kc) {
    v16bf pa[4];
#pragma unroll
    for (int mi = 0; mi < 4; ++mi)
      pa[mi] = load_a_frag(&P_s[wv][mi * 16][kc * 32], 72, lane);
    v16bf vf[2];
#pragma unroll
    for (int dj = 0; dj < 2; ++dj)
      vf[dj] = load_bt_frag(vh + (size_t)dj * 16 * NTOK + kc * 32, NTOK, lane);
#pragma unroll
    for (int mi = 0; mi < 4; ++mi)
#pragma unroll
      for (int dj = 0; dj < 2; ++dj)
        O[mi][dj] = wmma_bf16(pa[mi], vf[dj], O[mi][dj]);
  }

  // Write attention output in [token][channel] order for the final GEMM.
  {
    __bf16* ab = aout + ((size_t)b * NTOK + rowoff) * DIM + h * HD + coloff;
#pragma unroll
    for (int mi = 0; mi < 4; ++mi)
#pragma unroll
      for (int dj = 0; dj < 2; ++dj)
#pragma unroll
        for (int r = 0; r < 8; ++r)
          ab[(mi * 16 + r) * DIM + dj * 16] = to_bf16(O[mi][dj][r]);
  }
}

// ---------------------------------------------------------------------------
extern "C" void kernel_launch(void* const* d_in, const int* in_sizes, int n_in,
                              void* d_out, int out_size, void* d_ws, size_t ws_size,
                              hipStream_t stream) {
  (void)in_sizes; (void)n_in; (void)out_size; (void)ws_size;
  const float* x    = (const float*)d_in[0];
  const float* KV   = (const float*)d_in[1];
  const float* mask = (const float*)d_in[2];
  const float* Wq   = (const float*)d_in[3];
  const float* bq   = (const float*)d_in[4];
  const float* Wkv  = (const float*)d_in[5];
  const float* bkv  = (const float*)d_in[6];
  const float* Wp   = (const float*)d_in[7];
  const float* bp   = (const float*)d_in[8];
  const float* w1   = (const float*)d_in[9];
  const float* b1   = (const float*)d_in[10];
  const float* w2   = (const float*)d_in[11];
  const float* b2   = (const float*)d_in[12];
  const float* tau  = (const float*)d_in[13];
  const float* lri  = (const float*)d_in[14];
  float* out = (float*)d_out;

  const size_t M = (size_t)BWIN * NTOK;          // 131072 tokens
  const size_t CHUNK = M * DIM * sizeof(__bf16); // 64 MiB per bf16 tensor
  char* ws = (char*)d_ws;
  __bf16* q_bf  = (__bf16*)(ws + 0 * CHUNK);
  __bf16* k_bf  = (__bf16*)(ws + 1 * CHUNK);
  __bf16* v_bf  = (__bf16*)(ws + 2 * CHUNK);
  __bf16* ao_bf = (__bf16*)(ws + 3 * CHUNK);
  char*   tail  = ws + 4 * CHUNK;
  float2* combo = (float2*)tail;                         // 256 KB
  __bf16* wtq   = (__bf16*)(tail + (256 << 10));         // 128 KB
  __bf16* wtkv  = (__bf16*)(tail + (384 << 10));         // 256 KB
  __bf16* wtp   = (__bf16*)(tail + (640 << 10));         // 128 KB

  prep_kernel<<<16, 256, 0, stream>>>(w1, b1, w2, b2, tau, lri, combo);
  wtrans_kernel<<<256, 256, 0, stream>>>(Wq,  wtq,  256);
  wtrans_kernel<<<512, 256, 0, stream>>>(Wkv, wtkv, 512);
  wtrans_kernel<<<256, 256, 0, stream>>>(Wp,  wtp,  256);

  gemm_bf16_kernel<0, false><<<dim3(4, 1024), 256, 0, stream>>>(
      x, nullptr, wtq, bq, q_bf, nullptr, nullptr);

  gemm_bf16_kernel<1, false><<<dim3(8, 1024), 256, 0, stream>>>(
      KV, nullptr, wtkv, bkv, k_bf, v_bf, nullptr);

  attn_kernel<<<BWIN * 2, 128, 0, stream>>>(
      q_bf, k_bf, v_bf, mask, combo, ao_bf);

  gemm_bf16_kernel<2, true><<<dim3(4, 1024), 256, 0, stream>>>(
      nullptr, ao_bf, wtp, bp, nullptr, nullptr, out);
}
